// SimpleGCNLayer_45097156608542
// MI455X (gfx1250) — matrix operand order repaired and checked by reading, not compiled, not run on hardware
//
#include <hip/hip_runtime.h>
#include <hip/hip_bf16.h>

typedef _Float16 half4  __attribute__((ext_vector_type(4)));
typedef _Float16 half8  __attribute__((ext_vector_type(8)));
typedef _Float16 half16 __attribute__((ext_vector_type(16)));
typedef float    v8f    __attribute__((ext_vector_type(8)));

#define N_NODES 8192
#define FDIM    256   // F_IN == F_OUT == 256
#define BM      64    // rows per workgroup
#define KC      64    // K-chunk staged in LDS

// ---------------- kernel 1: dis[i] = rsqrt(rowsum(adj[i,:]) + 1e-9) ----------------
__global__ __launch_bounds__(256) void gcn_rowsum(const float* __restrict__ adj,
                                                  float* __restrict__ dis) {
    __shared__ float red[256];
    const int row = blockIdx.x;
    const int t   = threadIdx.x;
    const float* r = adj + (size_t)row * N_NODES;
    float s = 0.f;
#pragma unroll
    for (int j = 0; j < 8; ++j) {               // 8192 floats = 2048 float4 / 256 thr
        float4 v = ((const float4*)r)[t + 256 * j];
        s += v.x + v.y + v.z + v.w;
    }
    red[t] = s;
    __syncthreads();
    for (int off = 128; off > 0; off >>= 1) {
        if (t < off) red[t] += red[t + off];
        __syncthreads();
    }
    if (t == 0) dis[row] = rsqrtf(red[0] + 1e-9f);
}

// ---------------- kernel 2: xsT[f][i] = f16(dis[i] * x[i][f]) ----------------------
__global__ __launch_bounds__(256) void gcn_prep_x(const float* __restrict__ x,
                                                  const float* __restrict__ dis,
                                                  _Float16* __restrict__ xsT) {
    const int i = blockIdx.x * 256 + threadIdx.x;
    const int f = blockIdx.y;
    xsT[(size_t)f * N_NODES + i] = (_Float16)(dis[i] * x[(size_t)i * FDIM + f]);
}

// ---------------- kernel 3: Wf16 = f16(W), row-major [out][in] ---------------------
__global__ __launch_bounds__(256) void gcn_prep_w(const float* __restrict__ W,
                                                  _Float16* __restrict__ Wh) {
    const int idx = blockIdx.x * 256 + threadIdx.x;
    Wh[idx] = (_Float16)W[idx];
}

// ---------------- kernel 4: h = (adj @ xs) scaled by dis rows ----------------------
// WMMA f16 16x16x32, f32 accumulate. A staged (f32->f16) and B staged in LDS.
__global__ __launch_bounds__(256) void gcn_gemm1(const float* __restrict__ adj,
                                                 const _Float16* __restrict__ xsT,
                                                 const float* __restrict__ dis,
                                                 float* __restrict__ h) {
    __shared__ alignas(64) _Float16 ldsA[BM * KC];    // 8 KB
    __shared__ alignas(64) _Float16 ldsB[FDIM * KC];  // 32 KB
    const int t    = threadIdx.x;
    const int wave = t >> 5;
    const int lane = t & 31;
    const int m0 = blockIdx.x * BM;
    const int r  = (wave & 3) * 16;     // 16-row block within BM
    const int nb = (wave >> 2) * 128;   // feature half
    const int ml = lane & 15;
    const int hi = lane >> 4;

    v8f acc[8] = {};

    for (int kc = 0; kc < N_NODES; kc += KC) {
        // stage A: adj[m0..m0+63][kc..kc+63], convert f32 -> f16
#pragma unroll
        for (int j = 0; j < 4; ++j) {
            int lin = t + 256 * j;
            int row = lin >> 4;
            int c4  = (lin & 15) << 2;
            float4 v = *(const float4*)(adj + (size_t)(m0 + row) * N_NODES + kc + c4);
            half4 hv = {(_Float16)v.x, (_Float16)v.y, (_Float16)v.z, (_Float16)v.w};
            *(half4*)(&ldsA[row * KC + c4]) = hv;
        }
        // stage B: xsT[0..255][kc..kc+63] (already f16, contiguous b128 copies)
#pragma unroll
        for (int j = 0; j < 8; ++j) {
            int lin = t + 256 * j;
            int row = lin >> 3;
            int c8  = (lin & 7) << 3;
            *(uint4*)(&ldsB[row * KC + c8]) =
                *(const uint4*)(xsT + (size_t)row * N_NODES + kc + c8);
        }
        __syncthreads();

#pragma unroll
        for (int ks = 0; ks < KC / 32; ++ks) {
            // A fragment: lane holds row M=ml, K = hi*16 + 0..15 contiguous
            const int abase = (r + ml) * KC + ks * 32 + hi * 16;
            half8 alo = *(const half8*)(&ldsA[abase]);
            half8 ahi = *(const half8*)(&ldsA[abase + 8]);
            half16 a;
#pragma unroll
            for (int e = 0; e < 8; ++e) { a[e] = alo[e]; a[e + 8] = ahi[e]; }
#pragma unroll
            for (int nt = 0; nt < 8; ++nt) {
                // B fragment: lane holds column N=ml, K = hi*16 + 0..15 contiguous
                const int bbase = (nb + nt * 16 + ml) * KC + ks * 32 + hi * 16;
                half16 b = *(const half16*)(&ldsB[bbase]);
                acc[nt] = __builtin_amdgcn_wmma_f32_16x16x32_f16(
                    false, a, false, b, (short)0, acc[nt], false, false);
            }
        }
        __syncthreads();
    }

    // epilogue: C layout M = v + 8*hi, N = ml; scale by dis[m]
    float dv[8];
#pragma unroll
    for (int v = 0; v < 8; ++v) dv[v] = dis[m0 + r + v + 8 * hi];
#pragma unroll
    for (int nt = 0; nt < 8; ++nt) {
        const int n = nb + nt * 16 + ml;
#pragma unroll
        for (int v = 0; v < 8; ++v) {
            const int m = m0 + r + v + 8 * hi;
            h[(size_t)m * FDIM + n] = acc[nt][v] * dv[v];
        }
    }
}

// ---------------- kernel 5: out = relu(h @ W^T + b), in place over h ---------------
__global__ __launch_bounds__(256) void gcn_gemm2(const float* __restrict__ h,
                                                 const _Float16* __restrict__ Wh,
                                                 const float* __restrict__ bias,
                                                 float* __restrict__ out) {
    const int t    = threadIdx.x;
    const int wave = t >> 5;
    const int lane = t & 31;
    const int m0 = blockIdx.x * BM;
    const int r  = (wave & 3) * 16;
    const int nb = (wave >> 2) * 128;
    const int ml = lane & 15;
    const int hi = lane >> 4;

    v8f acc[8] = {};
    const float* arow = h + (size_t)(m0 + r + ml) * FDIM;

#pragma unroll
    for (int ks = 0; ks < FDIM / 32; ++ks) {
        const int k = ks * 32 + hi * 16;
        float4 a0 = *(const float4*)(arow + k);
        float4 a1 = *(const float4*)(arow + k + 4);
        float4 a2 = *(const float4*)(arow + k + 8);
        float4 a3 = *(const float4*)(arow + k + 12);
        half16 a = {(_Float16)a0.x, (_Float16)a0.y, (_Float16)a0.z, (_Float16)a0.w,
                    (_Float16)a1.x, (_Float16)a1.y, (_Float16)a1.z, (_Float16)a1.w,
                    (_Float16)a2.x, (_Float16)a2.y, (_Float16)a2.z, (_Float16)a2.w,
                    (_Float16)a3.x, (_Float16)a3.y, (_Float16)a3.z, (_Float16)a3.w};
#pragma unroll
        for (int nt = 0; nt < 8; ++nt) {
            // B[k][n] = W[n][k]; W row-major [out][in] -> contiguous 32B per lane
            const int nrow = nb + nt * 16 + ml;
            half16 b = *(const half16*)(Wh + (size_t)nrow * FDIM + ks * 32 + hi * 16);
            acc[nt] = __builtin_amdgcn_wmma_f32_16x16x32_f16(
                false, a, false, b, (short)0, acc[nt], false, false);
        }
    }
    __syncthreads();  // all waves finish reading h (aliases out) before any store
#pragma unroll
    for (int nt = 0; nt < 8; ++nt) {
        const int n  = nb + nt * 16 + ml;
        const float bn = bias[n];
#pragma unroll
        for (int v = 0; v < 8; ++v) {
            const int m = m0 + r + v + 8 * hi;
            float o = acc[nt][v] + bn;
            out[(size_t)m * FDIM + n] = o > 0.f ? o : 0.f;
        }
    }
}

extern "C" void kernel_launch(void* const* d_in, const int* in_sizes, int n_in,
                              void* d_out, int out_size, void* d_ws, size_t ws_size,
                              hipStream_t stream) {
    (void)in_sizes; (void)n_in; (void)out_size; (void)ws_size;
    const float* x   = (const float*)d_in[0];
    const float* adj = (const float*)d_in[1];
    const float* W   = (const float*)d_in[2];
    const float* b   = (const float*)d_in[3];
    float* out = (float*)d_out;

    char* ws = (char*)d_ws;
    _Float16* xsT = (_Float16*)ws;                                       // 4 MB
    _Float16* Wh  = (_Float16*)(ws + (size_t)N_NODES * FDIM * 2);        // 128 KB
    float*    dis = (float*)(ws + (size_t)N_NODES * FDIM * 2 + FDIM * FDIM * 2); // 32 KB

    gcn_rowsum<<<N_NODES, 256, 0, stream>>>(adj, dis);
    dim3 gx(N_NODES / 256, FDIM);
    gcn_prep_x<<<gx, 256, 0, stream>>>(x, dis, xsT);
    gcn_prep_w<<<(FDIM * FDIM) / 256, 256, 0, stream>>>(W, Wh);
    gcn_gemm1<<<N_NODES / BM, 256, 0, stream>>>(adj, xsT, dis, out);   // h -> d_out
    gcn_gemm2<<<N_NODES / BM, 256, 0, stream>>>(out, Wh, b, out);      // in-place
}
